// MultiHeadSelfAttentionBlock_12120397710040
// MI455X (gfx1250) — matrix operand order repaired and checked by reading, not compiled, not run on hardware
//
#include <hip/hip_runtime.h>
#include <hip/hip_bf16.h>

// MI455X / gfx1250: wave32, WMMA 16x16x32 bf16, f32 accumulate.
// Strategy: one fp32->bf16 pass, QKV GEMM (WMMA) -> Q,K scaled/bias, V stored
// transposed; flash-attention with transposed-score trick (per-lane softmax
// state); output GEMM (WMMA) -> fp32 + bias.

typedef __attribute__((ext_vector_type(16))) __bf16        v16bf;
typedef __attribute__((ext_vector_type(8)))  float         v8f;
typedef __attribute__((ext_vector_type(8)))  unsigned int  v8u;

#define B_  4
#define S_  2048
#define E_  1024
#define H_  16
#define HD_ 64

static __device__ __forceinline__ unsigned f2bf1(float x) {
    unsigned u = __builtin_bit_cast(unsigned, x);
    return (u + 0x7fffu + ((u >> 16) & 1u)) >> 16;   // RNE bf16
}
static __device__ __forceinline__ unsigned pack2bf(float lo, float hi) {
    return f2bf1(lo) | (f2bf1(hi) << 16);
}
static __device__ __forceinline__ v16bf asbf(v8u v) {
    return __builtin_bit_cast(v16bf, v);
}
static __device__ __forceinline__ v8f v8f_zero() {
    v8f z;
#pragma unroll
    for (int i = 0; i < 8; ++i) z[i] = 0.f;
    return z;
}

// ---------------------------------------------------------------- convert ---
__global__ void cvt_bf16(const float* __restrict__ src,
                         unsigned short* __restrict__ dst, long n) {
    long i      = (long)blockIdx.x * blockDim.x + threadIdx.x;
    long stride = (long)gridDim.x * blockDim.x;
    for (; i < n; i += stride) dst[i] = (unsigned short)f2bf1(src[i]);
}

// ------------------------------------------------------------- GEMM core ----
// Per-wave 32x64 output tile: A (MxK bf16 row-major), W (NxK bf16 row-major,
// i.e. B = W^T). K in steps of 32, 8 WMMAs per step.
static __device__ __forceinline__ void gemm_core(
    const unsigned short* __restrict__ A,
    const unsigned short* __restrict__ W,
    int m0, int n0, int K, v8f (&c)[2][4]) {
    int lane = threadIdx.x & 31;
    int lm = lane & 15, hh = lane >> 4;
    int akoff[8];
#pragma unroll
    for (int j = 0; j < 8; ++j) akoff[j] = ((j >= 4) ? 16 : 0) + 8 * hh + 2 * (j & 3);

    const unsigned short* a0p = A + (long)(m0 + lm) * K;
    const unsigned short* a1p = A + (long)(m0 + 16 + lm) * K;
    const unsigned short* wp[4];
#pragma unroll
    for (int fi = 0; fi < 4; ++fi) wp[fi] = W + (long)(n0 + fi * 16 + lm) * K;

#pragma unroll
    for (int r = 0; r < 2; ++r)
#pragma unroll
        for (int fi = 0; fi < 4; ++fi) c[r][fi] = v8f_zero();

    for (int k0 = 0; k0 < K; k0 += 32) {
        v8u a0, a1;
#pragma unroll
        for (int j = 0; j < 8; ++j) {
            a0[j] = *(const unsigned*)(a0p + k0 + akoff[j]);
            a1[j] = *(const unsigned*)(a1p + k0 + akoff[j]);
        }
        v16bf A0 = asbf(a0), A1 = asbf(a1);
#pragma unroll
        for (int fi = 0; fi < 4; ++fi) {
            v8u bb;
#pragma unroll
            for (int j = 0; j < 8; ++j)
                bb[j] = *(const unsigned*)(wp[fi] + k0 + akoff[j]);
            v16bf Bv = asbf(bb);
            c[0][fi] = __builtin_amdgcn_wmma_f32_16x16x32_bf16(
                false, A0, false, Bv, (short)0, c[0][fi], false, false);
            c[1][fi] = __builtin_amdgcn_wmma_f32_16x16x32_bf16(
                false, A1, false, Bv, (short)0, c[1][fi], false, false);
        }
    }
}

// ------------------------------------------------------------- QKV GEMM -----
// qkv[m,f] = x[m,:] . W_qkv[f,:] + b_qkv[f];  f = head*192 + c*64 + d
// Q scaled by 1/sqrt(HD)=0.125 and stored [b,h,s,d]; K stored [b,h,s,d];
// V stored transposed [b,h,d,s].
__global__ void qkv_gemm(const unsigned short* __restrict__ xb,
                         const unsigned short* __restrict__ Wb,
                         const float* __restrict__ bias,
                         unsigned short* __restrict__ Qb,
                         unsigned short* __restrict__ Kb,
                         unsigned short* __restrict__ Vt) {
    int wid = blockIdx.x * (blockDim.x >> 5) + (threadIdx.x >> 5);
    const int mtiles = (B_ * S_) / 32;              // 256
    int mt = wid % mtiles, nt = wid / mtiles;       // nt < 48
    int m0 = mt * 32, n0 = nt * 64;
    v8f c[2][4];
    gemm_core(xb, Wb, m0, n0, E_, c);

    int lane = threadIdx.x & 31, lm = lane & 15, hh = lane >> 4;
#pragma unroll
    for (int fi = 0; fi < 4; ++fi) {
        int f     = n0 + fi * 16 + lm;
        int head  = f / (3 * HD_);
        int rem   = f - head * 3 * HD_;
        int ctype = rem >> 6;
        int d     = rem & 63;
        float bia = bias[f];
        float scl = (ctype == 0) ? 0.125f : 1.0f;
        unsigned short* dst = (ctype == 0) ? Qb : ((ctype == 1) ? Kb : Vt);
#pragma unroll
        for (int r = 0; r < 2; ++r)
#pragma unroll
            for (int j = 0; j < 8; ++j) {
                int m = m0 + r * 16 + j + 8 * hh;
                int b = m >> 11;            // / 2048
                int s = m & 2047;
                float v = (c[r][fi][j] + bia) * scl;
                long idx = (ctype == 2)
                               ? ((((long)b * H_ + head) * HD_ + d) * S_ + s)
                               : ((((long)b * H_ + head) * S_ + s) * HD_ + d);
                dst[idx] = (unsigned short)f2bf1(v);
            }
    }
}

// ------------------------------------------------------- flash attention ----
// One wave per 16-query tile. Computes S^T = K * Q^T so each lane owns one
// query column -> softmax state is per-lane scalar. 8 WMMAs per 32-kv step.
__global__ void attn(const unsigned short* __restrict__ Qb,
                     const unsigned short* __restrict__ Kb,
                     const unsigned short* __restrict__ Vt,
                     unsigned short* __restrict__ Ob,
                     const int* __restrict__ is_causal) {
    const int causal = is_causal[0];
    int wid = blockIdx.x * (blockDim.x >> 5) + (threadIdx.x >> 5);
    int qt = wid & 127;             // S/16 = 128 tiles
    int bh = wid >> 7;              // b*H + h, < 64
    int q0 = qt * 16;
    int lane = threadIdx.x & 31, lm = lane & 15, hh = lane >> 4;
    int akoff[8];
#pragma unroll
    for (int j = 0; j < 8; ++j) akoff[j] = ((j >= 4) ? 16 : 0) + 8 * hh + 2 * (j & 3);

    // Q^T B-fragments (K-dim = head dim, N = query), preloaded once.
    const unsigned short* qrow = Qb + ((long)bh * S_ + q0 + lm) * HD_;
    v16bf qf[2];
#pragma unroll
    for (int kk = 0; kk < 2; ++kk) {
        v8u t;
#pragma unroll
        for (int j = 0; j < 8; ++j)
            t[j] = *(const unsigned*)(qrow + kk * 32 + akoff[j]);
        qf[kk] = asbf(t);
    }

    v8f o[4];                       // O^T accumulators (64 head dims x 16 q)
#pragma unroll
    for (int dt = 0; dt < 4; ++dt) o[dt] = v8f_zero();
    float mrun = -1e30f, lsum = 0.f;

    const unsigned short* kbase = Kb + (long)bh * S_ * HD_;
    const unsigned short* vtb   = Vt + (long)bh * HD_ * S_;

    int kv_end = causal ? (q0 + 16) : S_;
    for (int kv0 = 0; kv0 < kv_end; kv0 += 32) {
        // S^T tile: rows = 32 kv positions (2 M-tiles), cols = 16 queries
        v8f sc[2];
#pragma unroll
        for (int mt = 0; mt < 2; ++mt) {
            v8f acc = v8f_zero();
#pragma unroll
            for (int kk = 0; kk < 2; ++kk) {
                v8u t;
                const unsigned short* kr =
                    kbase + (long)(kv0 + mt * 16 + lm) * HD_ + kk * 32;
#pragma unroll
                for (int j = 0; j < 8; ++j)
                    t[j] = *(const unsigned*)(kr + akoff[j]);
                acc = __builtin_amdgcn_wmma_f32_16x16x32_bf16(
                    false, asbf(t), false, qf[kk], (short)0, acc, false, false);
            }
            sc[mt] = acc;
        }

        // causal mask (wave-uniform branch; only trailing blocks hit it)
        if (causal && (kv0 + 31 > q0)) {
            int qg = q0 + lm;
#pragma unroll
            for (int j = 0; j < 8; ++j) {
                int kg = kv0 + j + 8 * hh;
                if (kg > qg)      sc[0][j] = -1e30f;
                if (kg + 16 > qg) sc[1][j] = -1e30f;
            }
        }

        // online softmax: each lane owns one query column
        float pm = sc[0][0];
#pragma unroll
        for (int j = 1; j < 8; ++j) pm = fmaxf(pm, sc[0][j]);
#pragma unroll
        for (int j = 0; j < 8; ++j) pm = fmaxf(pm, sc[1][j]);
        pm = fmaxf(pm, __shfl_xor(pm, 16, 32));
        float mnew  = fmaxf(mrun, pm);
        float alpha = __expf(mrun - mnew);
        mrun = mnew;
        float psum = 0.f;
#pragma unroll
        for (int j = 0; j < 8; ++j) { sc[0][j] = __expf(sc[0][j] - mnew); psum += sc[0][j]; }
#pragma unroll
        for (int j = 0; j < 8; ++j) { sc[1][j] = __expf(sc[1][j] - mnew); psum += sc[1][j]; }
        psum += __shfl_xor(psum, 16, 32);
        lsum = lsum * alpha + psum;
#pragma unroll
        for (int dt = 0; dt < 4; ++dt)
#pragma unroll
            for (int j = 0; j < 8; ++j) o[dt][j] *= alpha;

        // repack P^T (C-layout) into B-fragment layout: 8 packs + 8 shuffles
        unsigned pc0[4], pc1[4], xc0[4], xc1[4];
#pragma unroll
        for (int i = 0; i < 4; ++i) {
            pc0[i] = pack2bf(sc[0][2 * i], sc[0][2 * i + 1]);
            pc1[i] = pack2bf(sc[1][2 * i], sc[1][2 * i + 1]);
            xc0[i] = (unsigned)__shfl_xor((int)pc0[i], 16, 32);
            xc1[i] = (unsigned)__shfl_xor((int)pc1[i], 16, 32);
        }
        v8u bp;
#pragma unroll
        for (int j = 0; j < 4; ++j) bp[j] = hh ? xc1[j] : pc0[j];
#pragma unroll
        for (int j = 4; j < 8; ++j) bp[j] = hh ? pc1[j - 4] : xc0[j - 4];
        v16bf Bp = asbf(bp);

        // O^T += V^T * P^T   (V^T contiguous thanks to transposed V store)
#pragma unroll
        for (int dt = 0; dt < 4; ++dt) {
            v8u t;
            const unsigned short* vr = vtb + (long)(dt * 16 + lm) * S_ + kv0;
#pragma unroll
            for (int j = 0; j < 8; ++j)
                t[j] = *(const unsigned*)(vr + 16 * hh + 2 * j);
            o[dt] = __builtin_amdgcn_wmma_f32_16x16x32_bf16(
                false, asbf(t), false, Bp, (short)0, o[dt], false, false);
        }
    }

    // normalize and scatter O -> [b, s, h*64+d] (bf16 for out-proj GEMM)
    float rl = 1.0f / lsum;
    int b = bh >> 4, head = bh & 15;
    long obase = ((long)b * S_ + q0 + lm) * E_ + head * HD_;
#pragma unroll
    for (int dt = 0; dt < 4; ++dt)
#pragma unroll
        for (int j = 0; j < 8; ++j) {
            int d = dt * 16 + j + 8 * hh;
            Ob[obase + d] = (unsigned short)f2bf1(o[dt][j] * rl);
        }
}

// -------------------------------------------------------- output GEMM -------
__global__ void out_gemm(const unsigned short* __restrict__ Ob,
                         const unsigned short* __restrict__ Wob,
                         const float* __restrict__ bias,
                         float* __restrict__ out) {
    int wid = blockIdx.x * (blockDim.x >> 5) + (threadIdx.x >> 5);
    const int mtiles = (B_ * S_) / 32;              // 256
    int mt = wid % mtiles, nt = wid / mtiles;       // nt < 16
    int m0 = mt * 32, n0 = nt * 64;
    v8f c[2][4];
    gemm_core(Ob, Wob, m0, n0, E_, c);

    int lane = threadIdx.x & 31, lm = lane & 15, hh = lane >> 4;
#pragma unroll
    for (int fi = 0; fi < 4; ++fi) {
        int f = n0 + fi * 16 + lm;
        float bia = bias[f];
#pragma unroll
        for (int r = 0; r < 2; ++r)
#pragma unroll
            for (int j = 0; j < 8; ++j) {
                int m = m0 + r * 16 + j + 8 * hh;
                out[(long)m * E_ + f] = c[r][fi][j] + bia;
            }
    }
}

// ----------------------------------------------------------------- launch ---
extern "C" void kernel_launch(void* const* d_in, const int* in_sizes, int n_in,
                              void* d_out, int out_size, void* d_ws, size_t ws_size,
                              hipStream_t stream) {
    const float* x     = (const float*)d_in[0];
    const float* W_qkv = (const float*)d_in[1];
    const float* b_qkv = (const float*)d_in[2];
    const float* W_out = (const float*)d_in[3];
    const float* b_out = (const float*)d_in[4];
    const int* is_causal = (const int*)d_in[5];

    const long n_x  = (long)B_ * S_ * E_;     // 8388608
    const long n_Wq = 3L * E_ * E_;           // 3145728
    const long n_Wo = (long)E_ * E_;          // 1048576

    char* ws = (char*)d_ws;
    unsigned short* xb  = (unsigned short*)(ws);
    unsigned short* Wb  = (unsigned short*)(ws + 16777216);
    unsigned short* Wob = (unsigned short*)(ws + 23068672);
    unsigned short* Qb  = (unsigned short*)(ws + 25165824);
    unsigned short* Kb  = (unsigned short*)(ws + 41943040);
    unsigned short* Vt  = (unsigned short*)(ws + 58720256);
    unsigned short* Ob  = (unsigned short*)(ws + 75497472);   // end 92274688 B

    cvt_bf16<<<2048, 256, 0, stream>>>(x, xb, n_x);
    cvt_bf16<<<1024, 256, 0, stream>>>(W_qkv, Wb, n_Wq);
    cvt_bf16<<<512,  256, 0, stream>>>(W_out, Wob, n_Wo);

    // 256 m-tiles * 48 n-tiles = 12288 wave-tiles, 4 waves/block
    qkv_gemm<<<3072, 128, 0, stream>>>(xb, Wb, b_qkv, Qb, Kb, Vt);

    // 64 (b,h) * 128 q-tiles = 8192 waves, 8 waves/block
    attn<<<1024, 256, 0, stream>>>(Qb, Kb, Vt, Ob, is_causal);

    // 256 m-tiles * 16 n-tiles = 4096 wave-tiles, 4 waves/block
    out_gemm<<<1024, 128, 0, stream>>>(Ob, Wob, b_out, (float*)d_out);
}
